// PatchRNNBlock_36438502539381
// MI455X (gfx1250) — compile-verified, matrix operand order checked by reading, not run on hardware
//
#include <hip/hip_runtime.h>
#include <hip/hip_bf16.h>
#include <math.h>

// ---------------------------------------------------------------------------
// PatchRNNBlock for MI455X (gfx1250, wave32, WMMA 16x16x32 f16)
//   B=2048, IN=512, K=16, T=96, H=8, L=497
// Pipeline:
//   1) embed_kernel : p = gelu((window(x)+pe) @ Wseg^T + bseg)  -> ws (f16)
//   2) stats_kernel : per-l batchnorm scale/bias over (B,K)
//   3) lstm0_kernel : bi-LSTM layer 0 (wave0=fwd, wave1=bwd), o0 -> ws (f16)
//   4) lstm1_kernel : bi-LSTM layer 1, captures final states + o1[:,L-1]
//   5) fb_kernel    : fb = LN(concat) * g + b + xe[:,L-1,:]
//   6) decode_kernel: 96 autoregressive steps, 4 WMMA cells + WMMA MLP
// Activations are branch-free, built from native v_exp_f32 / v_rcp_f32
// (ocml tanhf/erff are divergent 2-path polynomials -> poison for the
// latency-bound recurrent inner loops).
// ---------------------------------------------------------------------------

typedef _Float16 half_t;
typedef __attribute__((ext_vector_type(16))) _Float16 v16h;
typedef __attribute__((ext_vector_type(8)))  _Float16 v8h;
typedef __attribute__((ext_vector_type(8)))  float    v8f;

static constexpr int B_  = 2048;
static constexpr int IN_ = 512;
static constexpr int K_  = 16;
static constexpr int T_  = 96;
static constexpr int H_  = 8;
static constexpr int L_  = IN_ - K_ + 1;   // 497
static constexpr int NTILES = B_ / 16;     // 128

#define DEVI __device__ __forceinline__

DEVI float fast_exp2(float x)
{
#if __has_builtin(__builtin_amdgcn_exp2f)
    return __builtin_amdgcn_exp2f(x);      // single v_exp_f32
#else
    return exp2f(x);
#endif
}
DEVI float fast_rcp(float x)
{
#if __has_builtin(__builtin_amdgcn_rcpf)
    return __builtin_amdgcn_rcpf(x);       // single v_rcp_f32
#else
    return 1.0f / x;
#endif
}
// branch-free tanh: 1 - 2/(exp(2x)+1); exact saturation at +-1, no NaN path
DEVI float fast_tanh(float x)
{
#if __has_builtin(__builtin_amdgcn_tanh_f32)
    return __builtin_amdgcn_tanh_f32(x);   // hardware V_TANH_F32 if available
#else
    const float e = fast_exp2(x * 2.8853900817779268f);   // exp(2x)
    return 1.0f - 2.0f * fast_rcp(e + 1.0f);
#endif
}
// branch-free sigmoid: 1/(1+exp(-x))
DEVI float sigf(float x)
{
    return fast_rcp(1.0f + fast_exp2(x * -1.4426950408889634f));
}
// tanh-form GELU (branch-free)
DEVI float geluf(float x)
{
    const float u = 0.7978845608028654f * __builtin_fmaf(0.044715f * x, x * x, x);
    return 0.5f * x * (1.0f + fast_tanh(u));
}

// ---------------------------------------------------------------------------
// Build B fragment for one LSTM gate-pair from Wih (32x16) / Whh (32x8).
// Gates = [x_t(16) | h(8) | 0(8)] (K=32) x B(32x16).
// B[k][n]: k<16 -> Wih[n][k]; 16<=k<24 -> Whh[n][k-16]; else 0.
// f16 B layout (mirror of A): lanes 0-15 hold K=0..7 (elems 0-7) and
// K=16..23 (elems 8-15); lanes 16-31 hold K=8..15 and K=24..31; n = lane&15.
// ---------------------------------------------------------------------------
DEVI void load_cellB(const float* __restrict__ Wih, const float* __restrict__ Whh,
                     int lane, v16h& Blo, v16h& Bhi)
{
    const int n  = lane & 15;
    const int hi = lane >> 4;
#pragma unroll
    for (int q = 0; q < 8; ++q) {
        if (hi == 0) {
            Blo[q]     = (half_t)Wih[n * 16 + q];
            Blo[8 + q] = (half_t)Whh[n * 8 + q];
            Bhi[q]     = (half_t)Wih[(n + 16) * 16 + q];
            Bhi[8 + q] = (half_t)Whh[(n + 16) * 8 + q];
        } else {
            Blo[q]     = (half_t)Wih[n * 16 + 8 + q];
            Blo[8 + q] = (half_t)0.f;
            Bhi[q]     = (half_t)Wih[(n + 16) * 16 + 8 + q];
            Bhi[8 + q] = (half_t)0.f;
        }
    }
}

// ---------------------------------------------------------------------------
// One LSTM cell step for a 16-row batch tile, one wave.
//  xa     : per-lane A fragment: input features kb..kb+7 of row m=lane&15
//  hstage : LDS [16 rows][8 cols] f16, holds current h (read for A, updated)
//  c[8]   : f32 cell state, replicated in lane pairs (j, j+8)
// gates 16x32 = A(16x32) x [Blo|Bhi] -> D0 = [i|f], D1 = [g|o]
// ---------------------------------------------------------------------------
DEVI void lstm_step(const v16h& Blo, const v16h& Bhi, half_t* hstage,
                    const v8h& xa, float c[8], int lane)
{
    const int  hi  = lane >> 4;
    const int  m   = lane & 15;
    const int  j   = lane & 7;
    const bool low = (m < 8);

    v16h A;
#pragma unroll
    for (int q = 0; q < 8; ++q) A[q] = xa[q];
#pragma unroll
    for (int q = 0; q < 8; ++q) A[8 + q] = (hi == 0) ? hstage[m * 8 + q] : (half_t)0.f;

    v8f z = {};
    v8f d0 = __builtin_amdgcn_wmma_f32_16x16x32_f16(false, A, false, Blo, (short)0, z, false, false);
    v8f d1 = __builtin_amdgcn_wmma_f32_16x16x32_f16(false, A, false, Bhi, (short)0, z, false, false);

#pragma unroll
    for (int v = 0; v < 8; ++v) {
        float pa = __shfl_xor(d0[v], 8, 32);
        float pb = __shfl_xor(d1[v], 8, 32);
        float ig = low ? d0[v] : pa;
        float fg = low ? pa    : d0[v];
        float gg = low ? d1[v] : pb;
        float og = low ? pb    : d1[v];
        float cn = sigf(fg) * c[v] + sigf(ig) * fast_tanh(gg);
        float hn = sigf(og) * fast_tanh(cn);
        c[v] = cn;
        if (low) hstage[(v + 8 * hi) * 8 + j] = (half_t)hn;
    }
}

// ---------------------------------------------------------------------------
// Kernel 1: patch embed. p[r, n] = gelu(sum_k (x[b,l+k]+pe[l,k]) * Wseg[n,k] + bseg[n])
// r = b*L + l. One WMMA per 16-row tile.
// ---------------------------------------------------------------------------
__global__ void embed_kernel(const float* __restrict__ x, const float* __restrict__ pe,
                             const float* __restrict__ Wseg, const float* __restrict__ bseg,
                             half_t* __restrict__ p_ws)
{
    const int lane = threadIdx.x & 31;
    const int wave = threadIdx.x >> 5;
    const int n    = lane & 15;
    const int hi   = lane >> 4;
    const int kb   = hi * 8;

    v16h Bf;
#pragma unroll
    for (int q = 0; q < 8; ++q) {
        Bf[q]     = (half_t)Wseg[n * 16 + kb + q];
        Bf[8 + q] = (half_t)0.f;
    }
    const float bn = bseg[n];

    const int ntile = (B_ * L_) / 16;
    for (int tb = blockIdx.x * 8 + wave; tb < ntile; tb += gridDim.x * 8) {
        const int r = tb * 16 + n;          // this lane's A-row
        const int b = r / L_;
        const int l = r - b * L_;
        v16h A;
#pragma unroll
        for (int q = 0; q < 8; ++q) {
            A[q]     = (half_t)(x[(size_t)b * IN_ + l + kb + q] + pe[l * 16 + kb + q]);
            A[8 + q] = (half_t)0.f;
        }
        v8f z = {};
        v8f d = __builtin_amdgcn_wmma_f32_16x16x32_f16(false, A, false, Bf, (short)0, z, false, false);
#pragma unroll
        for (int v = 0; v < 8; ++v) {
            const int r2 = tb * 16 + v + 8 * hi;
            p_ws[(size_t)r2 * 16 + n] = (half_t)geluf(d[v] + bn);
        }
    }
}

// ---------------------------------------------------------------------------
// Kernel 2: per-l batchnorm statistics over (B, K); emit scale/bias.
// ---------------------------------------------------------------------------
__global__ void stats_kernel(const half_t* __restrict__ p_ws,
                             const float* __restrict__ g_bn, const float* __restrict__ b_bn,
                             float* __restrict__ scale, float* __restrict__ bias)
{
    __shared__ float ssum[256], ssq[256];
    const int l = blockIdx.x;
    float s = 0.f, q = 0.f;
    for (int i = threadIdx.x; i < B_ * 16; i += blockDim.x) {
        const int b = i >> 4, k = i & 15;
        const float v = (float)p_ws[((size_t)b * L_ + l) * 16 + k];
        s += v; q += v * v;
    }
    ssum[threadIdx.x] = s; ssq[threadIdx.x] = q;
    __syncthreads();
    for (int off = 128; off > 0; off >>= 1) {
        if ((int)threadIdx.x < off) {
            ssum[threadIdx.x] += ssum[threadIdx.x + off];
            ssq[threadIdx.x]  += ssq[threadIdx.x + off];
        }
        __syncthreads();
    }
    if (threadIdx.x == 0) {
        const float inv_n = 1.0f / (float)(B_ * 16);
        const float mu  = ssum[0] * inv_n;
        const float var = ssq[0] * inv_n - mu * mu;
        const float inv = rsqrtf(var + 1e-5f);
        const float g   = g_bn[l];
        scale[l] = inv * g;
        bias[l]  = b_bn[l] - mu * inv * g;
    }
}

// ---------------------------------------------------------------------------
// Kernel 3: bi-LSTM layer 0. Block = 64 threads: wave0 forward, wave1 backward.
// o0_ws row layout: [b][t][0..7]=fwd h, [8..15]=bwd h (f16).
// states: [idx][B][8] f32; idx 0..3 = hf0,cf0,hb0,cb0
// ---------------------------------------------------------------------------
__global__ void lstm0_kernel(const half_t* __restrict__ p_ws,
                             const float* __restrict__ scale, const float* __restrict__ bias,
                             const float* __restrict__ Wih_f, const float* __restrict__ Whh_f,
                             const float* __restrict__ Wih_b, const float* __restrict__ Whh_b,
                             half_t* __restrict__ o0_ws, float* __restrict__ states)
{
    __shared__ __align__(16) half_t hstage_s[2][16 * 8];
    const int lane = threadIdx.x & 31;
    const int dir  = threadIdx.x >> 5;
    half_t* hstage = hstage_s[dir];
    const int b0 = blockIdx.x * 16;
    const int hi = lane >> 4, m = lane & 15, j = lane & 7, kb = hi * 8;

    v16h Blo, Bhi;
    load_cellB(dir ? Wih_b : Wih_f, dir ? Whh_b : Whh_f, lane, Blo, Bhi);

    if (lane < 16) {
#pragma unroll
        for (int q = 0; q < 8; ++q) hstage[lane * 8 + q] = (half_t)0.f;
    }
    float c[8];
#pragma unroll
    for (int v = 0; v < 8; ++v) c[v] = 0.f;

    for (int s = 0; s < L_; ++s) {
        const int t = dir ? (L_ - 1 - s) : s;
        const float sc = scale[t], bi = bias[t];
        const half_t* prow = p_ws + ((size_t)(b0 + m) * L_ + t) * 16 + kb;
        v8h ph = *(const v8h*)prow;
        if (s + 1 < L_) {
            const int tn = dir ? (t - 1) : (t + 1);
            __builtin_prefetch(p_ws + ((size_t)(b0 + m) * L_ + tn) * 16, 0, 1);
        }
        v8h xa;
#pragma unroll
        for (int q = 0; q < 8; ++q) xa[q] = (half_t)((float)ph[q] * sc + bi);
        lstm_step(Blo, Bhi, hstage, xa, c, lane);
        if (lane < 16) {
            v8h hv;
#pragma unroll
            for (int q = 0; q < 8; ++q) hv[q] = hstage[m * 8 + q];
            *(v8h*)(o0_ws + ((size_t)(b0 + m) * L_ + t) * 16 + dir * 8) = hv;
        }
    }
    float* hout = states + (size_t)(dir * 2 + 0) * B_ * 8;
    float* cout = states + (size_t)(dir * 2 + 1) * B_ * 8;
    if (m < 8) {
#pragma unroll
        for (int v = 0; v < 8; ++v) {
            const int row = v + 8 * hi;
            hout[(size_t)(b0 + row) * 8 + j] = (float)hstage[row * 8 + j];
            cout[(size_t)(b0 + row) * 8 + j] = c[v];
        }
    }
}

// ---------------------------------------------------------------------------
// Kernel 4: bi-LSTM layer 1 over o0. states idx 4..7 = hf1,cf1,hb1,cb1.
// Captures of1[:,L-1] (fwd final) and ob1[:,L-1] (first bwd step).
// ---------------------------------------------------------------------------
__global__ void lstm1_kernel(const half_t* __restrict__ o0_ws,
                             const float* __restrict__ Wih_f, const float* __restrict__ Whh_f,
                             const float* __restrict__ Wih_b, const float* __restrict__ Whh_b,
                             float* __restrict__ states,
                             float* __restrict__ of1_last, float* __restrict__ ob1_last)
{
    __shared__ __align__(16) half_t hstage_s[2][16 * 8];
    const int lane = threadIdx.x & 31;
    const int dir  = threadIdx.x >> 5;
    half_t* hstage = hstage_s[dir];
    const int b0 = blockIdx.x * 16;
    const int hi = lane >> 4, m = lane & 15, j = lane & 7, kb = hi * 8;

    v16h Blo, Bhi;
    load_cellB(dir ? Wih_b : Wih_f, dir ? Whh_b : Whh_f, lane, Blo, Bhi);

    if (lane < 16) {
#pragma unroll
        for (int q = 0; q < 8; ++q) hstage[lane * 8 + q] = (half_t)0.f;
    }
    float c[8];
#pragma unroll
    for (int v = 0; v < 8; ++v) c[v] = 0.f;

    for (int s = 0; s < L_; ++s) {
        const int t = dir ? (L_ - 1 - s) : s;
        v8h xa = *(const v8h*)(o0_ws + ((size_t)(b0 + m) * L_ + t) * 16 + kb);
        if (s + 1 < L_) {
            const int tn = dir ? (t - 1) : (t + 1);
            __builtin_prefetch(o0_ws + ((size_t)(b0 + m) * L_ + tn) * 16, 0, 1);
        }
        lstm_step(Blo, Bhi, hstage, xa, c, lane);
        if (dir == 1 && s == 0 && m < 8) {   // ob1[:, L-1]
#pragma unroll
            for (int v = 0; v < 8; ++v) {
                const int row = v + 8 * hi;
                ob1_last[(size_t)(b0 + row) * 8 + j] = (float)hstage[row * 8 + j];
            }
        }
    }
    float* hout = states + (size_t)(4 + dir * 2) * B_ * 8;
    float* cout = states + (size_t)(5 + dir * 2) * B_ * 8;
    if (m < 8) {
#pragma unroll
        for (int v = 0; v < 8; ++v) {
            const int row = v + 8 * hi;
            const float hv = (float)hstage[row * 8 + j];
            hout[(size_t)(b0 + row) * 8 + j] = hv;
            cout[(size_t)(b0 + row) * 8 + j] = c[v];
            if (dir == 0) of1_last[(size_t)(b0 + row) * 8 + j] = hv;  // of1[:, L-1]
        }
    }
}

// ---------------------------------------------------------------------------
// Kernel 5: fb = LN(concat(of1_last, ob1_last)) * g_ln + b_ln + xe[:, L-1, :]
// ---------------------------------------------------------------------------
__global__ void fb_kernel(const float* __restrict__ of1_last, const float* __restrict__ ob1_last,
                          const float* __restrict__ g_ln, const float* __restrict__ b_ln,
                          const half_t* __restrict__ p_ws,
                          const float* __restrict__ scale, const float* __restrict__ bias,
                          float* __restrict__ fb)
{
    const int b = blockIdx.x * blockDim.x + threadIdx.x;
    if (b >= B_) return;
    float o[16];
#pragma unroll
    for (int k = 0; k < 8; ++k) {
        o[k]     = of1_last[(size_t)b * 8 + k];
        o[8 + k] = ob1_last[(size_t)b * 8 + k];
    }
    float mu = 0.f;
#pragma unroll
    for (int k = 0; k < 16; ++k) mu += o[k];
    mu *= 0.0625f;
    float var = 0.f;
#pragma unroll
    for (int k = 0; k < 16; ++k) { const float d = o[k] - mu; var += d * d; }
    var *= 0.0625f;
    const float inv = rsqrtf(var + 1e-5f);
    const float sc = scale[L_ - 1], bi = bias[L_ - 1];
#pragma unroll
    for (int k = 0; k < 16; ++k) {
        const float xe = (float)p_ws[((size_t)b * L_ + (L_ - 1)) * 16 + k] * sc + bi;
        fb[(size_t)b * 16 + k] = (o[k] - mu) * inv * g_ln[k] + b_ln[k] + xe;
    }
}

// ---------------------------------------------------------------------------
// Kernel 6: decoder. One wave per 16-batch tile runs all T=96 steps.
// Per step: 4 WMMA LSTM cells, per-row LN, WMMA MLP 16->512 (W1 in LDS),
// branch-free GELU, shuffle-reduced 512->1 projection.
// ---------------------------------------------------------------------------
__global__ void decode_kernel(const float* __restrict__ states, const float* __restrict__ fb,
                              const float* __restrict__ Wih_f0, const float* __restrict__ Whh_f0,
                              const float* __restrict__ Wih_b0, const float* __restrict__ Whh_b0,
                              const float* __restrict__ Wih_f1, const float* __restrict__ Whh_f1,
                              const float* __restrict__ Wih_b1, const float* __restrict__ Whh_b1,
                              const float* __restrict__ g_ln, const float* __restrict__ b_ln,
                              const float* __restrict__ W1, const float* __restrict__ b1,
                              const float* __restrict__ W2, const float* __restrict__ b2,
                              float* __restrict__ out)
{
    __shared__ __align__(16) half_t W1lds[512 * 16];
    __shared__ float b1lds[512], w2lds[512];
    __shared__ __align__(16) half_t hbuf[4][16 * 8];   // f0, b0, f1, b1
    __shared__ float obuf[16 * 16];

    const int lane = threadIdx.x;      // 32 threads / block
    const int hi = lane >> 4, m = lane & 15, j = lane & 7, kb = hi * 8;
    const int b0 = blockIdx.x * 16;

    for (int i = lane; i < 512 * 16; i += 32) W1lds[i] = (half_t)W1[i];
    for (int i = lane; i < 512; i += 32) { b1lds[i] = b1[i]; w2lds[i] = W2[i]; }

    v16h Bf0l, Bf0h, Bb0l, Bb0h, Bf1l, Bf1h, Bb1l, Bb1h;
    load_cellB(Wih_f0, Whh_f0, lane, Bf0l, Bf0h);
    load_cellB(Wih_b0, Whh_b0, lane, Bb0l, Bb0h);
    load_cellB(Wih_f1, Whh_f1, lane, Bf1l, Bf1h);
    load_cellB(Wih_b1, Whh_b1, lane, Bb1l, Bb1h);

    // init h (LDS, f16) and c (regs, f32) from encoder final states
#pragma unroll
    for (int cell = 0; cell < 4; ++cell) {
        const float* hsrc = states + (size_t)(cell * 2) * B_ * 8;
        if (lane < 16) {
#pragma unroll
            for (int q = 0; q < 8; ++q)
                hbuf[cell][lane * 8 + q] = (half_t)hsrc[(size_t)(b0 + lane) * 8 + q];
        }
    }
    float cf0[8], cb0[8], cf1[8], cb1[8];
    auto ldc = [&](float (&cd)[8], int sidx) {
        const float* csrc = states + (size_t)sidx * B_ * 8;
#pragma unroll
        for (int v = 0; v < 8; ++v) cd[v] = csrc[(size_t)(b0 + v + 8 * hi) * 8 + j];
    };
    ldc(cf0, 1); ldc(cb0, 3); ldc(cf1, 5); ldc(cb1, 7);

    float in_row[16];
#pragma unroll
    for (int k = 0; k < 16; ++k) in_row[k] = fb[(size_t)(b0 + m) * 16 + k];

    for (int t = 0; t < T_; ++t) {
        v8h xa;
#pragma unroll
        for (int q = 0; q < 8; ++q) xa[q] = (half_t)in_row[kb + q];
        lstm_step(Bf0l, Bf0h, hbuf[0], xa, cf0, lane);
        lstm_step(Bb0l, Bb0h, hbuf[1], xa, cb0, lane);

        v8h ma;   // m0 = [hf0 | hb0]
#pragma unroll
        for (int q = 0; q < 8; ++q)
            ma[q] = (hi == 0) ? hbuf[0][m * 8 + q] : hbuf[1][m * 8 + q];
        lstm_step(Bf1l, Bf1h, hbuf[2], ma, cf1, lane);
        lstm_step(Bb1l, Bb1h, hbuf[3], ma, cb1, lane);

        // o = concat(hf1, hb1) -> obuf (raw, f32)
        if (lane < 16) {
#pragma unroll
            for (int q = 0; q < 8; ++q) {
                obuf[lane * 16 + q]     = (float)hbuf[2][lane * 8 + q];
                obuf[lane * 16 + 8 + q] = (float)hbuf[3][lane * 8 + q];
            }
        }
#pragma unroll
        for (int k = 0; k < 16; ++k) in_row[k] = obuf[m * 16 + k];

        // LayerNorm over 16 features (per row, computed redundantly per lane)
        float mu = 0.f;
#pragma unroll
        for (int k = 0; k < 16; ++k) mu += in_row[k];
        mu *= 0.0625f;
        float var = 0.f;
#pragma unroll
        for (int k = 0; k < 16; ++k) { const float d = in_row[k] - mu; var += d * d; }
        var *= 0.0625f;
        const float inv = rsqrtf(var + 1e-5f);

        v16h zA;
#pragma unroll
        for (int q = 0; q < 8; ++q) {
            zA[q]     = (half_t)((in_row[kb + q] - mu) * inv * g_ln[kb + q] + b_ln[kb + q]);
            zA[8 + q] = (half_t)0.f;
        }

        // MLP: h = gelu(z @ W1^T + b1); y = h @ W2^T + b2
        float yacc[8];
#pragma unroll
        for (int v = 0; v < 8; ++v) yacc[v] = 0.f;
        for (int nt = 0; nt < 32; ++nt) {
            v16h Bt;
            v8h w = *(const v8h*)(W1lds + ((nt * 16 + m) * 16 + kb));
#pragma unroll
            for (int q = 0; q < 8; ++q) { Bt[q] = w[q]; Bt[8 + q] = (half_t)0.f; }
            v8f z = {};
            v8f d = __builtin_amdgcn_wmma_f32_16x16x32_f16(false, zA, false, Bt, (short)0, z, false, false);
            const float bb = b1lds[nt * 16 + m];
            const float ww = w2lds[nt * 16 + m];
#pragma unroll
            for (int v = 0; v < 8; ++v) yacc[v] += geluf(d[v] + bb) * ww;
        }
        // reduce 16 lanes sharing the same row group
#pragma unroll
        for (int mask = 1; mask < 16; mask <<= 1) {
#pragma unroll
            for (int v = 0; v < 8; ++v) yacc[v] += __shfl_xor(yacc[v], mask, 32);
        }
        if (m == 0) {
#pragma unroll
            for (int v = 0; v < 8; ++v)
                out[(size_t)(b0 + hi * 8 + v) * T_ + t] = yacc[v] + b2[0];
        }
    }
}

// ---------------------------------------------------------------------------
extern "C" void kernel_launch(void* const* d_in, const int* in_sizes, int n_in,
                              void* d_out, int out_size, void* d_ws, size_t ws_size,
                              hipStream_t stream)
{
    const float* x      = (const float*)d_in[0];
    const float* pe     = (const float*)d_in[1];
    const float* Wseg   = (const float*)d_in[2];
    const float* bseg   = (const float*)d_in[3];
    const float* g_bn   = (const float*)d_in[4];
    const float* b_bn   = (const float*)d_in[5];
    const float* Wih_f0 = (const float*)d_in[6];
    const float* Whh_f0 = (const float*)d_in[7];
    const float* Wih_b0 = (const float*)d_in[8];
    const float* Whh_b0 = (const float*)d_in[9];
    const float* Wih_f1 = (const float*)d_in[10];
    const float* Whh_f1 = (const float*)d_in[11];
    const float* Wih_b1 = (const float*)d_in[12];
    const float* Whh_b1 = (const float*)d_in[13];
    const float* g_ln   = (const float*)d_in[14];
    const float* b_ln   = (const float*)d_in[15];
    const float* W1     = (const float*)d_in[16];
    const float* b1     = (const float*)d_in[17];
    const float* W2     = (const float*)d_in[18];
    const float* b2     = (const float*)d_in[19];

    char* ws = (char*)d_ws;
    size_t off = 0;
    auto alloc = [&](size_t bytes) {
        off = (off + 255) & ~(size_t)255;
        size_t o = off; off += bytes; return o;
    };
    half_t* p_ws   = (half_t*)(ws + alloc((size_t)B_ * L_ * 16 * sizeof(half_t)));
    half_t* o0_ws  = (half_t*)(ws + alloc((size_t)B_ * L_ * 16 * sizeof(half_t)));
    float*  scale  = (float*)(ws + alloc((size_t)L_ * sizeof(float)));
    float*  biasv  = (float*)(ws + alloc((size_t)L_ * sizeof(float)));
    float*  states = (float*)(ws + alloc((size_t)8 * B_ * 8 * sizeof(float)));
    float*  of1l   = (float*)(ws + alloc((size_t)B_ * 8 * sizeof(float)));
    float*  ob1l   = (float*)(ws + alloc((size_t)B_ * 8 * sizeof(float)));
    float*  fbv    = (float*)(ws + alloc((size_t)B_ * 16 * sizeof(float)));
    (void)ws_size; (void)in_sizes; (void)n_in; (void)out_size;

    embed_kernel<<<512, 256, 0, stream>>>(x, pe, Wseg, bseg, p_ws);
    stats_kernel<<<L_, 256, 0, stream>>>(p_ws, g_bn, b_bn, scale, biasv);
    lstm0_kernel<<<NTILES, 64, 0, stream>>>(p_ws, scale, biasv,
                                            Wih_f0, Whh_f0, Wih_b0, Whh_b0, o0_ws, states);
    lstm1_kernel<<<NTILES, 64, 0, stream>>>(o0_ws, Wih_f1, Whh_f1, Wih_b1, Whh_b1,
                                            states, of1l, ob1l);
    fb_kernel<<<(B_ + 255) / 256, 256, 0, stream>>>(of1l, ob1l, g_ln, b_ln,
                                                    p_ws, scale, biasv, fbv);
    decode_kernel<<<NTILES, 32, 0, stream>>>(states, fbv,
                                             Wih_f0, Whh_f0, Wih_b0, Whh_b0,
                                             Wih_f1, Whh_f1, Wih_b1, Whh_b1,
                                             g_ln, b_ln, W1, b1, W2, b2, (float*)d_out);
}
